// ChebGraphConv_46746424050357
// MI455X (gfx1250) — matrix operand order repaired and checked by reading, not compiled, not run on hardware
//
#include <hip/hip_runtime.h>

// ============================================================================
// ChebGraphConv for MI455X (gfx1250, wave32, WMMA + async-to-LDS).
//
// B=16, C_IN=32, T=64, N=1024, C_OUT=32, K=3 taps.
//   X1 = GSO @ X0 ; X2 = 2*GSO @ X1 - X0 ; out = sum_k Xk @ Wk + bias
// GSO shared across (b,t) -> each recursion step is ONE GEMM
//   [1024 x 1024] @ [1024 x 32768], 137 GFLOP vs ~0.4 GB -> compute bound
//   -> f16 WMMA 16x16x32 with f32 accumulate (values O(1), no overflow).
// New this round: double-buffered LDS tiles fed by GLOBAL_LOAD_ASYNC_TO_LDS
// (ASYNCcnt path, no VGPR round-trip) and XOR bank-swizzled fragment layout.
// ============================================================================

typedef __attribute__((ext_vector_type(16))) _Float16 v16h;
typedef __attribute__((ext_vector_type(8)))  float    v8f;

// ---- CDNA5 async copy: global -> LDS, 16B per lane, tracked by ASYNCcnt ----
__device__ __forceinline__ void async_copy_b128(unsigned lds_addr, const void* gptr) {
    asm volatile("global_load_async_to_lds_b128 %0, %1, off"
                 :: "v"(lds_addr), "v"(gptr) : "memory");
}
__device__ __forceinline__ void wait_async0() {
#if __has_builtin(__builtin_amdgcn_s_wait_asynccnt)
    __builtin_amdgcn_s_wait_asynccnt(0);
#else
    asm volatile("s_wait_asynccnt 0" ::: "memory");
#endif
}

// ---------------------------------------------------------------------------
// Fragment-layout addressing (CDNA5 ISA 16-bit A/B VGPR tables) for a
// 16x32 A-subtile / 32x16 B-subtile: each lane owns 32 contiguous bytes.
// The 16B half-select bit is XORed with lane[3] so a wave-wide ds_load_b128
// touches all 64 LDS banks (2-cycle optimum instead of 4-way conflicts).
//   A chunk (h, kg):  lane = h[3:0] + 16*k[3],  half = k[4]
//   B chunk (c, kg):  lane = c[3:0] + 16*k[3],  half = k[4]
// ---------------------------------------------------------------------------
__device__ __forceinline__ unsigned fragA_addr(int h, int kg) {
    unsigned L = (unsigned)((h & 15) + ((kg & 1) << 4));
    unsigned g = (unsigned)(kg >> 1);
    return (unsigned)((h >> 4) * 1024) + L * 32 + ((g << 4) ^ (((L >> 3) & 1) << 4));
}
__device__ __forceinline__ unsigned fragB_addr(int c, int kg) {
    unsigned L = (unsigned)((c & 15) + ((kg & 1) << 4));
    unsigned g = (unsigned)(kg >> 1);
    return (unsigned)((c >> 4) * 1024) + L * 32 + ((g << 4) ^ (((L >> 3) & 1) << 4));
}

// ---------------------------------------------------------------------------
// Kernel 0: gso fp32 -> f16, row-major [h][i].
// ---------------------------------------------------------------------------
__global__ __launch_bounds__(256)
void k_convert_gso(const float* __restrict__ g, _Float16* __restrict__ gh) {
    int idx = blockIdx.x * 256 + threadIdx.x;   // 1M elements
    gh[idx] = (_Float16)g[idx];
}

// ---------------------------------------------------------------------------
// Main Chebyshev GEMM: Out[c][h] = scale * sum_i G[h,i]*Bmat[c,i] (- X[c,h])
//   PASS 1: Bmat = x fp32 (converted in the pipeline), scale = 1
//   PASS 2: Bmat = Y1 f16 [c][node] (async-to-LDS), subtract X, scale = 2
// Block 256 = 8 waves; tile 128(h) x 128(c); K-step 32; double-buffered LDS.
// ---------------------------------------------------------------------------
template<int PASS>
__global__ __launch_bounds__(256)
void k_cheb_gemm(const _Float16* __restrict__ G,
                 const _Float16* __restrict__ Bh,
                 const float*    __restrict__ X,
                 _Float16*       __restrict__ Out)
{
    // buf0: A@0 B@8192 ; buf1: A@16384 B@24576 ; epilogue reuses 0..36863
    __shared__ __attribute__((aligned(32))) unsigned char smem[36864];

    const int tid  = threadIdx.x;
    const int wave = tid >> 5;
    const int lane = tid & 31;
    const int c0   = blockIdx.x * 128;
    const int h0   = blockIdx.y * 128;
    const int wh   = wave >> 2;   // 0..1 (h)
    const int wc   = wave & 3;    // 0..3 (c)
    const unsigned lds0 = (unsigned)(size_t)&smem[0];

    v8f acc[4][2];
    #pragma unroll
    for (int i = 0; i < 4; ++i)
        #pragma unroll
        for (int j = 0; j < 2; ++j) acc[i][j] = v8f{};

    float4 bp[4];   // PASS 1 staging pipeline registers (2 chunks x 8 floats)

    auto issueA = [&](int k0, unsigned base) {   // async: G tile, 2x16B/lane
        #pragma unroll
        for (int rep = 0; rep < 2; ++rep) {
            int q = tid + rep * 256, h = q >> 2, kg = q & 3;
            async_copy_b128(lds0 + base + fragA_addr(h, kg),
                            G + (size_t)(h0 + h) * 1024 + k0 + kg * 8);
        }
    };
    auto issueB2 = [&](int k0, unsigned base) {  // async: Y1 tile (PASS 2)
        #pragma unroll
        for (int rep = 0; rep < 2; ++rep) {
            int q = tid + rep * 256, c = q >> 2, kg = q & 3;
            async_copy_b128(lds0 + base + 8192u + fragB_addr(c, kg),
                            Bh + (size_t)(c0 + c) * 1024 + k0 + kg * 8);
        }
    };
    auto loadB1 = [&](int k0) {                  // PASS 1: x fp32 -> regs
        #pragma unroll
        for (int rep = 0; rep < 2; ++rep) {
            int q = tid + rep * 256, c = q >> 2, kg = q & 3;
            int cg = c0 + c, ci = cg & 31, bt = cg >> 5;
            int row = ((bt >> 6) * 32 + ci) * 64 + (bt & 63);
            const float4* p = (const float4*)(X + (size_t)row * 1024 + k0 + kg * 8);
            bp[rep * 2 + 0] = p[0];
            bp[rep * 2 + 1] = p[1];
        }
    };
    auto storeB1 = [&](unsigned base) {          // PASS 1: cvt + ds_store
        #pragma unroll
        for (int rep = 0; rep < 2; ++rep) {
            int q = tid + rep * 256, c = q >> 2, kg = q & 3;
            float4 f0 = bp[rep * 2 + 0], f1 = bp[rep * 2 + 1];
            __attribute__((aligned(16))) _Float16 hv[8] = {
                (_Float16)f0.x, (_Float16)f0.y, (_Float16)f0.z, (_Float16)f0.w,
                (_Float16)f1.x, (_Float16)f1.y, (_Float16)f1.z, (_Float16)f1.w};
            *(uint4*)(smem + base + 8192u + fragB_addr(c, kg)) = *(const uint4*)hv;
        }
    };
    auto consume = [&](unsigned base) {          // 6 frags, 8 WMMAs
        const unsigned sw = (unsigned)(((lane >> 3) & 1) << 4);
        v16h af[4], bf[2];
        #pragma unroll
        for (int ta = 0; ta < 4; ++ta) {
            unsigned a = base + (unsigned)((wh * 4 + ta) * 1024) + (unsigned)lane * 32;
            union { uint4 u[2]; v16h v; } r;
            r.u[0] = *(const uint4*)(smem + a + sw);
            r.u[1] = *(const uint4*)(smem + a + (16u ^ sw));
            af[ta] = r.v;
        }
        #pragma unroll
        for (int tb = 0; tb < 2; ++tb) {
            unsigned a = base + 8192u + (unsigned)((wc * 2 + tb) * 1024) + (unsigned)lane * 32;
            union { uint4 u[2]; v16h v; } r;
            r.u[0] = *(const uint4*)(smem + a + sw);
            r.u[1] = *(const uint4*)(smem + a + (16u ^ sw));
            bf[tb] = r.v;
        }
        #pragma unroll
        for (int ta = 0; ta < 4; ++ta)
            #pragma unroll
            for (int tb = 0; tb < 2; ++tb)
                acc[ta][tb] = __builtin_amdgcn_wmma_f32_16x16x32_f16(
                    false, af[ta], false, bf[tb], (short)0, acc[ta][tb], false, false);
    };

    // ---- software pipeline over 32 K-steps, double-buffered ---------------
    issueA(0, 0u);
    if (PASS == 2) issueB2(0, 0u);
    else { loadB1(0); storeB1(0u); }
    wait_async0();
    __syncthreads();

    #pragma unroll 1
    for (int it = 0; it < 32; ++it) {
        unsigned cur = (it & 1) ? 16384u : 0u;
        unsigned nxt = cur ^ 16384u;
        if (it < 31) {
            issueA((it + 1) * 32, nxt);          // overlaps with WMMAs below
            if (PASS == 2) issueB2((it + 1) * 32, nxt);
            else loadB1((it + 1) * 32);
        }
        consume(cur);
        if (it < 31 && PASS == 1) storeB1(nxt);
        if (it < 31) wait_async0();
        __syncthreads();
    }

    // ---- epilogue: re-tile via LDS so global stores are node-contiguous ---
    const float scale = (PASS == 1) ? 1.0f : 2.0f;
    {
        unsigned wbase = wave * 4608u;           // 32 rows x 144B per wave
        #pragma unroll
        for (int ta = 0; ta < 4; ++ta)
            #pragma unroll
            for (int tb = 0; tb < 2; ++tb) {
                int crow = tb * 16 + (lane & 15);
                #pragma unroll
                for (int r = 0; r < 8; ++r) {
                    int hcol = ta * 16 + r + ((lane >> 4) << 3);
                    *(_Float16*)(smem + wbase + crow * 144 + hcol * 2) =
                        (_Float16)(acc[ta][tb][r] * scale);
                }
            }
    }
    __syncthreads();
    {
        unsigned wbase = wave * 4608u;
        int c  = c0 + wc * 32 + lane;            // this lane's output row
        int hb = h0 + wh * 64;
        if (PASS == 2) {
            // Y2 = 2*G@Y1 - X : subtract x[c][h] fp32 on the fly
            int ci = c & 31, bt = c >> 5;
            const float* xrow = X + (size_t)(((bt >> 6) * 32 + ci) * 64 + (bt & 63)) * 1024 + hb;
            #pragma unroll
            for (int j = 0; j < 8; ++j) {
                uint4 u = *(const uint4*)(smem + wbase + lane * 144 + j * 16);
                _Float16* hv = (_Float16*)&u;
                float4 f0 = *(const float4*)(xrow + j * 8);
                float4 f1 = *(const float4*)(xrow + j * 8 + 4);
                float fs[8] = {f0.x, f0.y, f0.z, f0.w, f1.x, f1.y, f1.z, f1.w};
                #pragma unroll
                for (int e = 0; e < 8; ++e)
                    hv[e] = (_Float16)((float)hv[e] - fs[e]);
                *(uint4*)(Out + (size_t)c * 1024 + hb + j * 8) = u;
            }
        } else {
            #pragma unroll
            for (int j = 0; j < 8; ++j) {
                uint4 u = *(const uint4*)(smem + wbase + lane * 144 + j * 16);
                *(uint4*)(Out + (size_t)c * 1024 + hb + j * 8) = u;
            }
        }
    }
}

// ---------------------------------------------------------------------------
// Epilogue GEMM: out[(bt,h),co] = sum_{k,ci} Yk[(bt,ci),h]*W[k,ci,co] + bias
// K = 96 (3 taps x 32 ci), N-dim = 32 (2 co tiles). Memory bound (~330 MB);
// A-fragment gathers are 2B loads but coalesced across lanes (node contig).
// ---------------------------------------------------------------------------
__global__ __launch_bounds__(256)
void k_out_gemm(const float*    __restrict__ X,
                const _Float16* __restrict__ Y1,
                const _Float16* __restrict__ Y2,
                const float*    __restrict__ W,
                const float*    __restrict__ bias,
                float*          __restrict__ Out)
{
    const int tid  = threadIdx.x;
    const int wave = tid >> 5;
    const int lane = tid & 31;
    const int bt   = blockIdx.y;                 // 0..1023 (b*64+t)
    const int h0   = blockIdx.x * 128 + wave * 16;
    const int g    = lane >> 4;
    const int l    = lane & 15;

    // Weight B-fragments: lane = co + 16*ci[3], slots {ci[2:0], ci[4]*8}
    v16h wf[3][2];
    #pragma unroll
    for (int m = 0; m < 3; ++m)
        #pragma unroll
        for (int n = 0; n < 2; ++n)
            #pragma unroll
            for (int j = 0; j < 16; ++j) {
                int ci = ((j >> 3) << 4) + g * 8 + (j & 7);
                wf[m][n][j] = (_Float16)W[(m * 32 + ci) * 32 + n * 16 + l];
            }

    // A-fragments (row h = h0+l): lane = h + 16*ci[3], slots {ci[2:0], ci[4]*8}
    const int bb = bt >> 6, tt = bt & 63;
    v16h af[3];
    #pragma unroll
    for (int j = 0; j < 16; ++j) {
        int ci = ((j >> 3) << 4) + g * 8 + (j & 7);
        af[0][j] = (_Float16)X[(size_t)((bb * 32 + ci) * 64 + tt) * 1024 + h0 + l];
        af[1][j] = Y1[(size_t)(bt * 32 + ci) * 1024 + h0 + l];
        af[2][j] = Y2[(size_t)(bt * 32 + ci) * 1024 + h0 + l];
    }

    v8f acc[2] = {v8f{}, v8f{}};
    #pragma unroll
    for (int m = 0; m < 3; ++m)
        #pragma unroll
        for (int n = 0; n < 2; ++n)
            acc[n] = __builtin_amdgcn_wmma_f32_16x16x32_f16(
                false, af[m], false, wf[m][n], (short)0, acc[n], false, false);

    #pragma unroll
    for (int n = 0; n < 2; ++n) {
        float bv = bias[n * 16 + l];
        #pragma unroll
        for (int r = 0; r < 8; ++r) {
            int h = h0 + r + g * 8;
            Out[((size_t)bt * 1024 + h) * 32 + n * 16 + l] = acc[n][r] + bv;
        }
    }
}

// ---------------------------------------------------------------------------
// Launch. Workspace (requires ~130.1 MB of d_ws):
//   [0, 2MB) gso f16 | [2MB, 66MB) Y1 f16 [c][node] | [66MB, 130MB) Y2 f16
// ---------------------------------------------------------------------------
extern "C" void kernel_launch(void* const* d_in, const int* in_sizes, int n_in,
                              void* d_out, int out_size, void* d_ws, size_t ws_size,
                              hipStream_t stream) {
    const float* x    = (const float*)d_in[0];   // [16,32,64,1024] fp32
    const float* gso  = (const float*)d_in[1];   // [1024,1024] fp32
    const float* wgt  = (const float*)d_in[2];   // [3,32,32] fp32
    const float* bias = (const float*)d_in[3];   // [32] fp32
    float* out = (float*)d_out;                  // [16,64,1024,32] fp32

    char* ws = (char*)d_ws;
    _Float16* Gh = (_Float16*)ws;
    _Float16* Y1 = (_Float16*)(ws + (size_t)(2u << 20));
    _Float16* Y2 = (_Float16*)(ws + (size_t)(2u << 20) + (size_t)(64u << 20));

    k_convert_gso<<<4096, 256, 0, stream>>>(gso, Gh);

    dim3 g2(256, 8);   // 32768/128 c-tiles x 1024/128 h-tiles
    k_cheb_gemm<1><<<g2, 256, 0, stream>>>(Gh, (const _Float16*)nullptr, x, Y1);
    k_cheb_gemm<2><<<g2, 256, 0, stream>>>(Gh, Y1, x, Y2);

    dim3 g3(8, 1024);  // 1024/128 h-blocks x 1024 (b,t) pairs
    k_out_gemm<<<g3, 256, 0, stream>>>(x, Y1, Y2, wgt, bias, out);
}
